// ModulatedConv2d_25701084299423
// MI455X (gfx1250) — matrix operand order repaired and checked by reading, not compiled, not run on hardware
//
#include <hip/hip_runtime.h>

typedef __attribute__((ext_vector_type(16))) __bf16 v16bf;
typedef __attribute__((ext_vector_type(8)))  float  v8f;
typedef __attribute__((ext_vector_type(4)))  unsigned int v4u;
typedef __attribute__((ext_vector_type(8)))  int  v8i;
typedef __attribute__((ext_vector_type(4)))  int  v4i;

#define IN_C   512
#define OUT_C  512
#define B_     16
#define H_     64
#define W_     64
#define STYLE_DIM 512
#define EPS_F  1e-8f
#define SCALE_F 0.014731391f   /* 1/sqrt(512*9) */

__device__ __forceinline__ unsigned short f2bf(float f) {
    unsigned int u = __float_as_uint(f);
    u = (u + 0x7FFFu + ((u >> 16) & 1u)) >> 16;   // round-to-nearest-even
    return (unsigned short)u;
}

// s[b,ci] = style[b] . mod_w[ci] + mod_b[ci]
__global__ void style_mod_kernel(const float* __restrict__ style,
                                 const float* __restrict__ mod_w,
                                 const float* __restrict__ mod_b,
                                 float* __restrict__ s_out) {
    const int b  = blockIdx.x;
    const int ci = threadIdx.x;            // 512 threads
    const float* st = style + b * STYLE_DIM;
    const float* mw = mod_w + (size_t)ci * STYLE_DIM;
    float acc = 0.f;
#pragma unroll 8
    for (int j = 0; j < STYLE_DIM; ++j) acc += st[j] * mw[j];
    s_out[b * IN_C + ci] = acc + mod_b[ci];
}

// wsq[co,ci] = sum_tap weight[co,ci,tap]^2
__global__ void wsq_kernel(const float* __restrict__ weight,
                           float* __restrict__ wsq) {
    const int idx = blockIdx.x * blockDim.x + threadIdx.x;  // co*512+ci
    const float* w = weight + (size_t)idx * 9;
    float a = 0.f;
#pragma unroll
    for (int t = 0; t < 9; ++t) { const float v = w[t]; a += v * v; }
    wsq[idx] = a;
}

// demod[b,co] = rsqrt(SCALE^2 * sum_ci wsq[co,ci]*s[b,ci]^2 + eps)
__global__ void demod_kernel(const float* __restrict__ wsq,
                             const float* __restrict__ s_buf,
                             float* __restrict__ demod) {
    const int idx = blockIdx.x * blockDim.x + threadIdx.x;  // b*512+co
    const int b = idx >> 9, co = idx & 511;
    const float* wr = wsq  + (size_t)co * IN_C;
    const float* sr = s_buf + (size_t)b * IN_C;
    float acc = 0.f;
#pragma unroll 4
    for (int ci = 0; ci < IN_C; ++ci) { const float sv = sr[ci]; acc += wr[ci] * sv * sv; }
    demod[idx] = rsqrtf(SCALE_F * SCALE_F * acc + EPS_F);
}

// wT[tap][co][ci] = bf16(SCALE * weight[co][ci][tap])  (tap-major repack)
__global__ void pack_w_kernel(const float* __restrict__ weight,
                              unsigned short* __restrict__ wT) {
    const int idx = blockIdx.x * blockDim.x + threadIdx.x;  // (tap*512+co)*512+ci
    const int ci  = idx & 511;
    const int co  = (idx >> 9) & 511;
    const int tap = idx >> 18;
    const float v = SCALE_F * weight[((size_t)(co * IN_C + ci)) * 9 + tap];
    wT[idx] = f2bf(v);
}

// Implicit-GEMM 3x3 conv:
//  - block tile: 64 co x 128 pixels (two output rows)
//  - A: all 9 bf16 tap-tiles staged per ci-block by ONE Tensor Data Mover
//    descriptor (3D tile: 32 ci x 64 co x 9 taps), tracked by TENSORcnt
//  - B: modulated input halo (4 rows x 66 cols x 32 ci, bf16) staged once per
//    ci-block; all 9 taps read shifted fragments from it
//  - 8 waves as 2(M) x 4(N); each wave: 2 A-frags x 2 B-frags -> 4 WMMAs/tap,
//    36 v_wmma_f32_16x16x32_bf16 per wave per ci-block
#define XSTRIDE 72   /* padded col stride (16B aligned rows of 32 ci) */

__global__ __launch_bounds__(256)
void conv_wmma_kernel(const float* __restrict__ input,
                      const unsigned short* __restrict__ wT,
                      const float* __restrict__ s_buf,
                      const float* __restrict__ demod,
                      float* __restrict__ out) {
    const int h0  = blockIdx.x * 2;        // two output rows h0, h0+1
    const int coB = blockIdx.y * 64;       // co tile base
    const int b   = blockIdx.z;

    const int tid  = threadIdx.x;
    const int lane = tid & 31;
    const int wid  = tid >> 5;
    const int wm   = wid & 1;              // co sub-tile of 32
    const int wn   = wid >> 1;             // pixel sub-tile of 32 (0..3)

    __shared__ __align__(16) unsigned short aT[9 * 64 * 32];        // [tap][co][k]
    __shared__ __align__(16) unsigned short xT[4 * XSTRIDE * 32];   // [row][col][k]

    v8f acc00 = {}, acc01 = {}, acc10 = {}, acc11 = {};

    const float* xb = input + (size_t)b * IN_C * (H_ * W_);
    const float* sb = s_buf + (size_t)b * IN_C;

    // X staging: thread covers 9 cols x 4 rows of one ci
    const int x_ci  = tid >> 3;            // 0..31
    const int x_g   = (tid & 7) * 9;       // col group base 0..63

    // Fragment addresses (documented 16-bit WMMA layouts, wave32)
    const int nlo   = lane & 15;
    const int arow0 = wm * 32 + nlo;
    const int akb   = (lane < 16) ? 0 : 8;     // A: lanes 16-31 hold K=8..15 / 24..31
    const int bkb   = (lane < 16) ? 0 : 8;     // B: lanes 16-31 hold K=8..15 / 24..31
    const int r0    = wn >> 1;                 // which of the two output rows
    const int cb    = (wn * 32) & 63;          // col base within row (0 or 32)

#if __has_builtin(__builtin_amdgcn_tensor_load_to_lds)
    // Constant part of the TDM descriptor (ISA 8.3/8.4): 3D tile
    //   X = 32 ci (contiguous), Y = 64 co (stride 512), Z = 9 taps (stride 512*512)
    const unsigned lds_base = (unsigned)(size_t)(void*)aT;  // LDS byte offset
    v8i g1;
    g1[0] = 0x00010000;              // workgroup_mask=0, data_size=1 (2 bytes)
    g1[1] = (int)(512u << 16);       // tensor_dim0 = 512 (low 16)
    g1[2] = (int)(512u << 16);       // tensor_dim0 hi=0 | tensor_dim1 = 512 (low 16)
    g1[3] = (int)(32u << 16);        // tensor_dim1 hi=0 | tile_dim0 = 32
    g1[4] = (int)((9u << 16) | 64u); // tile_dim1 = 64, tile_dim2 = 9
    g1[5] = 512;                     // tensor_dim0_stride = 512 (low 32)
    g1[6] = 0;                       // dim0_stride hi | dim1_stride low16 (0x40000 -> 0)
    g1[7] = 4;                       // dim1_stride bits[47:16] = 0x40000 >> 16
    const v4i g2 = { 9, 0, 0, 0 };   // tensor_dim2 = 9
    const v4i g3 = { 0, 0, 0, 0 };
#else
    // Fallback A staging: thread covers 8 contiguous ci of one co row per tap
    const int a_row = tid >> 2;            // 0..63
    const int a_col = (tid & 3) * 8;       // 0,8,16,24
#endif

    for (int ci0 = 0; ci0 < IN_C; ci0 += 32) {
#if __has_builtin(__builtin_amdgcn_tensor_load_to_lds)
        // ---- stage all 9 A tap-tiles with one TDM descriptor (wave 0) ----
        if (wid == 0) {
            const unsigned long long ga =
                (unsigned long long)(size_t)wT + 2ull * ((size_t)coB * IN_C + ci0);
            v4u g0;
            g0[0] = 1u;                                  // count=1 (valid D#)
            g0[1] = lds_base;                            // lds_addr
            g0[2] = (unsigned)(ga & 0xFFFFFFFFu);        // global_addr[31:0]
            g0[3] = (unsigned)((ga >> 32) & 0x01FFFFFFu) // global_addr[56:32]
                    | (2u << 30);                        // type=2 ("image")
#if defined(__clang_major__) && (__clang_major__ >= 23)
            const v8i g4 = { 0, 0, 0, 0, 0, 0, 0, 0 };
            __builtin_amdgcn_tensor_load_to_lds(g0, g1, g2, g3, g4, 0);
#else
            __builtin_amdgcn_tensor_load_to_lds(g0, g1, g2, g3, 0);
#endif
        }
#else
#pragma unroll
        for (int tap = 0; tap < 9; ++tap) {
            *(uint4*)(aT + tap * (64 * 32) + a_row * 32 + a_col) =
                *(const uint4*)(wT + (size_t)(tap * OUT_C + coB + a_row) * IN_C
                                   + ci0 + a_col);
        }
#endif

        // ---- stage modulated input halo: rows h0-1..h0+2, cols -1..64 ----
        {
            const int   ci = ci0 + x_ci;
            const float sv = sb[ci];
            const float* xp = xb + (size_t)ci * (H_ * W_);
#pragma unroll
            for (int r = 0; r < 4; ++r) {
                const int  hh    = h0 + r - 1;
                const bool rowok = (hh >= 0) && (hh < H_);
                const int  hc    = rowok ? hh : 0;
                const float* xr  = xp + (size_t)hc * W_;
#pragma unroll
                for (int i = 0; i < 9; ++i) {
                    const int c = x_g + i;           // padded col 0..71
                    const int w = c - 1;
                    float v = 0.f;
                    if (rowok && w >= 0 && w < W_) v = xr[w] * sv;
                    xT[(r * XSTRIDE + c) * 32 + x_ci] = f2bf(v);
                }
            }
        }

#if __has_builtin(__builtin_amdgcn_tensor_load_to_lds)
        if (wid == 0) __builtin_amdgcn_s_wait_tensorcnt(0);  // TDM done -> LDS valid
#endif
        __syncthreads();

        // ---- 9 taps x 4 WMMAs, all operands from LDS ----
#pragma unroll
        for (int tap = 0; tap < 9; ++tap) {
            const int ky = tap / 3, kx = tap % 3;
            union Frag { v16bf v; uint4 q[2]; } fa0, fa1, fb0, fb1;
            const unsigned short* at = aT + tap * (64 * 32);
            fa0.q[0] = *(const uint4*)(at + arow0 * 32 + akb);
            fa0.q[1] = *(const uint4*)(at + arow0 * 32 + akb + 16);
            fa1.q[0] = *(const uint4*)(at + (arow0 + 16) * 32 + akb);
            fa1.q[1] = *(const uint4*)(at + (arow0 + 16) * 32 + akb + 16);

            const unsigned short* xt =
                xT + ((r0 + ky) * XSTRIDE + cb + nlo + kx) * 32;
            fb0.q[0] = *(const uint4*)(xt + bkb);
            fb0.q[1] = *(const uint4*)(xt + bkb + 16);
            fb1.q[0] = *(const uint4*)(xt + 16 * 32 + bkb);
            fb1.q[1] = *(const uint4*)(xt + 16 * 32 + bkb + 16);

            acc00 = __builtin_amdgcn_wmma_f32_16x16x32_bf16(
                        false, fa0.v, false, fb0.v, (short)0, acc00, false, false);
            acc01 = __builtin_amdgcn_wmma_f32_16x16x32_bf16(
                        false, fa0.v, false, fb1.v, (short)0, acc01, false, false);
            acc10 = __builtin_amdgcn_wmma_f32_16x16x32_bf16(
                        false, fa1.v, false, fb0.v, (short)0, acc10, false, false);
            acc11 = __builtin_amdgcn_wmma_f32_16x16x32_bf16(
                        false, fa1.v, false, fb1.v, (short)0, acc11, false, false);
        }
        __syncthreads();
    }

    // ---- epilogue: apply demod, scatter per documented C/D layout ----
    const int  mofs = (lane < 16) ? 0 : 8;
    const int  hrow = h0 + r0;
    float* obase = out + (size_t)b * OUT_C * (H_ * W_) + (size_t)hrow * W_;
#pragma unroll
    for (int j = 0; j < 8; ++j) {
        const int m0 = coB + wm * 32 + mofs + j;        // fa0 rows
        const int m1 = m0 + 16;                         // fa1 rows
        const float d0 = demod[b * OUT_C + m0];
        const float d1 = demod[b * OUT_C + m1];
        obase[(size_t)m0 * (H_ * W_) + cb + nlo]      = acc00[j] * d0;
        obase[(size_t)m0 * (H_ * W_) + cb + 16 + nlo] = acc01[j] * d0;
        obase[(size_t)m1 * (H_ * W_) + cb + nlo]      = acc10[j] * d1;
        obase[(size_t)m1 * (H_ * W_) + cb + 16 + nlo] = acc11[j] * d1;
    }
}

extern "C" void kernel_launch(void* const* d_in, const int* in_sizes, int n_in,
                              void* d_out, int out_size, void* d_ws, size_t ws_size,
                              hipStream_t stream) {
    const float* input  = (const float*)d_in[0];   // [16,512,64,64]
    const float* style  = (const float*)d_in[1];   // [16,512]
    const float* weight = (const float*)d_in[2];   // [512,512,3,3]
    const float* mod_w  = (const float*)d_in[3];   // [512,512]
    const float* mod_b  = (const float*)d_in[4];   // [512]
    float* out = (float*)d_out;                    // [16,512,64,64] fp32

    char* ws = (char*)d_ws;
    float*          s_buf = (float*)(ws);                       //  32 KB
    float*          demod = (float*)(ws + (32u << 10));         //  32 KB
    float*          wsq   = (float*)(ws + (64u << 10));         //   1 MB
    unsigned short* wT    = (unsigned short*)(ws + (64u << 10) + (1u << 20)); // 4.5 MB

    style_mod_kernel<<<dim3(B_), dim3(512), 0, stream>>>(style, mod_w, mod_b, s_buf);
    wsq_kernel<<<dim3((OUT_C * IN_C) / 256), dim3(256), 0, stream>>>(weight, wsq);
    demod_kernel<<<dim3((B_ * OUT_C) / 256), dim3(256), 0, stream>>>(wsq, s_buf, demod);
    pack_w_kernel<<<dim3((9 * OUT_C * IN_C) / 256), dim3(256), 0, stream>>>(weight, wT);

    conv_wmma_kernel<<<dim3(H_ / 2, OUT_C / 64, B_), dim3(256), 0, stream>>>(
        input, wT, s_buf, demod, out);
}
